// EarthSpecificBlock_90211493085838
// MI455X (gfx1250) — matrix operand order repaired and checked by reading, not compiled, not run on hardware
//
#include <hip/hip_runtime.h>

// ---------------- problem constants ----------------
#define L_TOK   147456     // 8*96*192 tokens
#define CDIM    256
#define NWIN    1024       // 4*16*16 windows
#define NTOK    144        // 2*6*12 tokens per window
#define QSZ     37748736   // 1024*8*144*32 elements for each of q,k,v
#define SCALE_Q 0.17677669529663687f   // 32^-0.5
#define EPSF    1e-5f

// ---------------- WMMA plumbing (gfx1250, wave32) ----------------
typedef __attribute__((ext_vector_type(16))) __bf16 v16bf;
typedef __attribute__((ext_vector_type(8)))  float  v8f;

union BFV { v16bf v; unsigned short u[16]; };

__device__ inline unsigned short f2bf(float f) {
  unsigned u = __float_as_uint(f);
  u += 0x7fffu + ((u >> 16) & 1u);        // round-to-nearest-even
  return (unsigned short)(u >> 16);
}

__device__ inline v8f wmma_bf16(v16bf a, v16bf b, v8f c) {
  // D = A(16x32) * B(32x16) + C, f32 accumulate
  return __builtin_amdgcn_wmma_f32_16x16x32_bf16(false, a, false, b, (short)0, c,
                                                 false, false);
}

// ---------------- CDNA5 async global->LDS (ASYNCcnt) ----------------
typedef int v4i_vs __attribute__((vector_size(16)));

__device__ inline void async_g2l_b128(const void* g, void* l) {
#if __has_builtin(__builtin_amdgcn_global_load_async_to_lds_b128)
  __builtin_amdgcn_global_load_async_to_lds_b128(
      (__attribute__((address_space(1))) v4i_vs*)g,
      (__attribute__((address_space(3))) v4i_vs*)l, 0, 0);
#else
  unsigned laddr = (unsigned)(unsigned long long)
      (__attribute__((address_space(3))) void*)l;
  asm volatile("global_load_async_to_lds_b128 %0, %1, off"
               :: "v"(laddr), "v"(g) : "memory");
#endif
}

template <int N>
__device__ inline void wait_async() {
#if __has_builtin(__builtin_amdgcn_s_wait_asynccnt)
  __builtin_amdgcn_s_wait_asynccnt(N);
#else
  asm volatile("s_wait_asynccnt %0" :: "n"(N) : "memory");
#endif
}

// A fragment: A[m][k] row-major, contiguous k order (k = 16*(lane/16) + s)
__device__ inline v16bf load_a_frag(const unsigned short* A, int lda) {
  int l = threadIdx.x & 31;
  int m = l & 15, khi = l >> 4;
  BFV r;
#pragma unroll
  for (int j = 0; j < 8; ++j) {
    int k = (khi << 4) + (j << 1);
    r.u[2*j]   = A[m*lda + k];
    r.u[2*j+1] = A[m*lda + k + 1];
  }
  return r.v;
}

// B fragment where B[k][n] = Wt[n][k] (weights stored N x K row-major)
__device__ inline v16bf load_b_frag_nk(const unsigned short* Wt, int ldw) {
  int l = threadIdx.x & 31;
  int n = l & 15, kb = (l >> 4) << 4;
  BFV r;
#pragma unroll
  for (int j = 0; j < 8; ++j) {
    r.u[2*j]   = Wt[n*ldw + kb + 2*j];
    r.u[2*j+1] = Wt[n*ldw + kb + 2*j + 1];
  }
  return r.v;
}

// B fragment where B stored [k][n] row-major
__device__ inline v16bf load_b_frag_kn(const unsigned short* B, int ldb) {
  int l = threadIdx.x & 31;
  int n = l & 15, kb = (l >> 4) << 4;
  BFV r;
#pragma unroll
  for (int j = 0; j < 8; ++j) {
    r.u[2*j]   = B[(kb + 2*j)*ldb + n];
    r.u[2*j+1] = B[(kb + 2*j + 1)*ldb + n];
  }
  return r.v;
}

// issue one K-step tile (A:128x32, B:64x32 bf16) via async copies; 3 ops/thread
__device__ inline void issue_gemm_tile(const unsigned short* __restrict__ gA,
                                       const unsigned short* __restrict__ gB,
                                       unsigned short* lA, unsigned short* lB,
                                       int tid) {
  int c = tid * 2;                       // A: 512 16B-chunks (row = c>>2, q = c&3)
  int row = c >> 2, q = c & 3;
  async_g2l_b128(gA + (long)row*CDIM + q*8, lA + row*40 + q*8);
  row = (c + 1) >> 2; q = (c + 1) & 3;
  async_g2l_b128(gA + (long)row*CDIM + q*8, lA + row*40 + q*8);
  row = tid >> 2; q = tid & 3;           // B: 256 chunks
  async_g2l_b128(gB + (long)row*CDIM + q*8, lB + row*40 + q*8);
}

// ---------------- prep kernels ----------------
__global__ __launch_bounds__(256) void k_f2bf(const float* __restrict__ s,
                                              unsigned short* __restrict__ d, int n) {
  int i = blockIdx.x * 256 + threadIdx.x;
  if (i < n) d[i] = f2bf(s[i]);
}

// bias8[h][n][m] gathered from bias_table[(pos*64+wti)*8 + h]
__global__ __launch_bounds__(256) void k_bias(const float* __restrict__ bt,
                                              const int* __restrict__ wtip,
                                              float* __restrict__ bias8) {
  int i = blockIdx.x * 256 + threadIdx.x;
  if (i >= 8 * NTOK * NTOK) return;
  int hh  = i / (NTOK * NTOK);
  int rem = i % (NTOK * NTOK);
  int n = rem / NTOK, m = rem % NTOK;
  int zi = n / 72, hi = (n / 12) % 6, wi = n % 12;
  int zj = m / 72, hj = (m / 12) % 6, wj = m % 12;
  int pos = (zi + 2*zj)*828 + (hi + 6*hj)*23 + (wi - wj + 11);
  bias8[i] = bt[(pos * 64 + wtip[0]) * 8 + hh];
}

// ---------------- LN1 + roll(-shift) + window partition -> bf16 ----------------
__global__ __launch_bounds__(256) void k_ln1_window(const float* __restrict__ x,
    const float* __restrict__ g, const float* __restrict__ b,
    unsigned short* __restrict__ xw) {
  int wv = threadIdx.x >> 5, l = threadIdx.x & 31;
  long o = (long)blockIdx.x * 8 + wv;            // token in window order
  int w = (int)(o / NTOK), n = (int)(o % NTOK);
  int wz = w >> 8, wh = (w >> 4) & 15, ww = w & 15;
  int lz = n / 72, lh = (n / 12) % 6, lw = n % 12;
  int gz = wz*2 + lz, gh = wh*6 + lh, gw = ww*12 + lw;
  int sz = gz + 1; if (sz >= 8)   sz -= 8;       // xs[g] = x[g + shift]
  int sh = gh + 3; if (sh >= 96)  sh -= 96;
  int sw = gw + 6; if (sw >= 192) sw -= 192;
  long src = ((long)(sz*96 + sh)*192 + sw) * CDIM;
  int c0 = l * 8;
  float4 a0 = *(const float4*)(x + src + c0);
  float4 a1 = *(const float4*)(x + src + c0 + 4);
  float vals[8] = {a0.x,a0.y,a0.z,a0.w,a1.x,a1.y,a1.z,a1.w};
  float s = 0.f, q = 0.f;
#pragma unroll
  for (int i = 0; i < 8; ++i) { s += vals[i]; q += vals[i]*vals[i]; }
#pragma unroll
  for (int off = 16; off > 0; off >>= 1) {
    s += __shfl_xor(s, off, 32);
    q += __shfl_xor(q, off, 32);
  }
  float mean = s * (1.f/256.f);
  float var  = q * (1.f/256.f) - mean*mean;
  float inv  = rsqrtf(var + EPSF);
  unsigned pk[4];
#pragma unroll
  for (int i = 0; i < 4; ++i) {
    float y0 = (vals[2*i]   - mean)*inv*g[c0+2*i]   + b[c0+2*i];
    float y1 = (vals[2*i+1] - mean)*inv*g[c0+2*i+1] + b[c0+2*i+1];
    pk[i] = (unsigned)f2bf(y0) | ((unsigned)f2bf(y1) << 16);
  }
  *(uint4*)(xw + o*CDIM + c0) = make_uint4(pk[0], pk[1], pk[2], pk[3]);
}

// ------- QKV GEMM: M=147456, K=256, N=768; async double-buffered staging -------
__global__ __launch_bounds__(256) void k_qkv(const unsigned short* __restrict__ xw,
    const unsigned short* __restrict__ wq, const float* __restrict__ bq,
    unsigned short* __restrict__ qkv) {
  __shared__ unsigned short At[2][128 * 40];
  __shared__ unsigned short Bt[2][64 * 40];
  int tid = threadIdx.x, wv = tid >> 5, l = tid & 31;
  long m0 = (long)blockIdx.x * 128;
  int  n0 = blockIdx.y * 64;
  const unsigned short* gA = xw + m0 * CDIM;
  const unsigned short* gB = wq + (long)n0 * CDIM;
  v8f acc[4] = {{0,0,0,0,0,0,0,0},{0,0,0,0,0,0,0,0},
                {0,0,0,0,0,0,0,0},{0,0,0,0,0,0,0,0}};
  issue_gemm_tile(gA, gB, At[0], Bt[0], tid);
  for (int kt = 0; kt < 8; ++kt) {
    int cur = kt & 1;
    if (kt < 7) {
      issue_gemm_tile(gA + (kt+1)*32, gB + (kt+1)*32, At[cur^1], Bt[cur^1], tid);
      wait_async<3>();           // this tile done; next tile (3 ops) in flight
    } else {
      wait_async<0>();
    }
    __syncthreads();
    v16bf a = load_a_frag(At[cur] + (wv*16)*40, 40);
#pragma unroll
    for (int j = 0; j < 4; ++j) {
      v16bf bf = load_b_frag_nk(Bt[cur] + (j*16)*40, 40);
      acc[j] = wmma_bf16(a, bf, acc[j]);
    }
    __syncthreads();             // protect buffer reuse
  }
#pragma unroll
  for (int j = 0; j < 4; ++j)
#pragma unroll
    for (int r = 0; r < 8; ++r) {
      long m  = m0 + wv*16 + ((l >> 4) << 3) + r;
      int col = n0 + j*16 + (l & 15);
      float v = acc[j][r] + bq[col];
      int which = col >> 8, rem = col & 255, head = rem >> 5, d = rem & 31;
      if (which == 0) v *= SCALE_Q;
      int w = (int)(m / NTOK), n = (int)(m % NTOK);
      long dst = (long)which * QSZ + (((long)(w*8 + head))*NTOK + n)*32 + d;
      qkv[dst] = f2bf(v);
    }
}

// ---------------- attention: one block per (window, head), 9 waves ----------------
__global__ __launch_bounds__(288) void k_attn(const unsigned short* __restrict__ qkv,
    const float* __restrict__ bias8, unsigned short* __restrict__ aout) {
  extern __shared__ char smem[];
  unsigned short* Ksm = (unsigned short*)smem;          // 144*40
  unsigned short* Vsm = Ksm + 144*40;                   // 160*32 (K padded to 160)
  unsigned short* Psm = Vsm + 160*32;                   // 144*168 (K padded to 160)
  int* regn = (int*)(Psm + 144*168);                    // 144 region ids
  int w = blockIdx.x, h = blockIdx.y, tid = threadIdx.x;
  const long base = ((long)(w*8 + h)) * NTOK * 32;
  const unsigned short* qg = qkv + base;
  const unsigned short* kg = qkv + QSZ + base;
  const unsigned short* vg = qkv + 2L*QSZ + base;
  // async-stage K and V (144 rows x 64B = 576 16B-chunks each)
  for (int c = tid; c < 576; c += 288) {
    int row = c >> 2, q = c & 3;
    async_g2l_b128(kg + row*32 + q*8, Ksm + row*40 + q*8);
    async_g2l_b128(vg + row*32 + q*8, Vsm + row*32 + q*8);
  }
  if (tid < 64) *(uint4*)(Vsm + 144*32 + tid*8) = make_uint4(0,0,0,0);
  if (tid < NTOK) {
    int n = tid;
    int lz = n/72, lh = (n/12)%6, lw = n%12;
    int wz = w>>8, wh = (w>>4)&15, ww = w&15;
    int gz = wz*2+lz, gh = wh*6+lh, gw = ww*12+lw;
    int cz = (gz < 6)  ? 0 : ((gz == 6)  ? 1 : 2);
    int ch = (gh < 90) ? 0 : ((gh < 93)  ? 1 : 2);
    int cw = (gw < 180)? 0 : ((gw < 186) ? 1 : 2);
    regn[n] = (cz*3 + ch)*3 + cw;
  }
  wait_async<0>();
  __syncthreads();

  int wv = tid >> 5, l = tid & 31;
  // Q strip fragment straight from global (each element used once)
  v16bf aq;
  {
    BFV r_;
    int m = wv*16 + (l & 15), khi = l >> 4;
#pragma unroll
    for (int j = 0; j < 8; ++j) {
      int k = (khi << 4) + (j << 1);
      r_.u[2*j]   = qg[m*32 + k];
      r_.u[2*j+1] = qg[m*32 + k + 1];
    }
    aq = r_.v;
  }
  v8f s[9];
#pragma unroll
  for (int jt = 0; jt < 9; ++jt) {
    v8f z = {0,0,0,0,0,0,0,0};
    v16bf bk = load_b_frag_nk(Ksm + (jt*16)*40, 40);    // B[d][m] = K[m][d]
    s[jt] = wmma_bf16(aq, bk, z);
  }
  // bias + shift mask
#pragma unroll
  for (int jt = 0; jt < 9; ++jt)
#pragma unroll
    for (int r = 0; r < 8; ++r) {
      int m = wv*16 + ((l >> 4) << 3) + r;
      int c = jt*16 + (l & 15);
      float v = s[jt][r] + bias8[((long)h*NTOK + m)*NTOK + c];
      if (regn[m] != regn[c]) v -= 100.f;
      s[jt][r] = v;
    }
  // row softmax (row spread over 16 lanes of one half-wave)
  float rm[8], rs[8];
#pragma unroll
  for (int r = 0; r < 8; ++r) {
    float mx = s[0][r];
#pragma unroll
    for (int jt = 1; jt < 9; ++jt) mx = fmaxf(mx, s[jt][r]);
    rm[r] = mx;
  }
#pragma unroll
  for (int off = 8; off > 0; off >>= 1)
#pragma unroll
    for (int r = 0; r < 8; ++r) rm[r] = fmaxf(rm[r], __shfl_xor(rm[r], off, 32));
#pragma unroll
  for (int r = 0; r < 8; ++r) rs[r] = 0.f;
#pragma unroll
  for (int jt = 0; jt < 9; ++jt)
#pragma unroll
    for (int r = 0; r < 8; ++r) {
      float e = __expf(s[jt][r] - rm[r]);
      s[jt][r] = e; rs[r] += e;
    }
#pragma unroll
  for (int off = 8; off > 0; off >>= 1)
#pragma unroll
    for (int r = 0; r < 8; ++r) rs[r] += __shfl_xor(rs[r], off, 32);
#pragma unroll
  for (int r = 0; r < 8; ++r) rs[r] = 1.f / rs[r];
  // P -> LDS (bf16), zero the K-pad columns 144..159
#pragma unroll
  for (int jt = 0; jt < 9; ++jt)
#pragma unroll
    for (int r = 0; r < 8; ++r) {
      int m = wv*16 + ((l >> 4) << 3) + r;
      Psm[m*168 + jt*16 + (l & 15)] = f2bf(s[jt][r] * rs[r]);
    }
#pragma unroll
  for (int r = 0; r < 8; ++r) {
    int m = wv*16 + ((l >> 4) << 3) + r;
    Psm[m*168 + 144 + (l & 15)] = 0;
  }
  __syncthreads();
  // out = P(16x160) @ V(160x32): 5 K-steps, 2 N-tiles
  v8f o0 = {0,0,0,0,0,0,0,0}, o1 = {0,0,0,0,0,0,0,0};
#pragma unroll
  for (int ks = 0; ks < 5; ++ks) {
    v16bf ap = load_a_frag(Psm + (wv*16)*168 + ks*32, 168);
    v16bf b0 = load_b_frag_kn(Vsm + (ks*32)*32 + 0, 32);
    v16bf b1 = load_b_frag_kn(Vsm + (ks*32)*32 + 16, 32);
    o0 = wmma_bf16(ap, b0, o0);
    o1 = wmma_bf16(ap, b1, o1);
  }
#pragma unroll
  for (int r = 0; r < 8; ++r) {
    int m = wv*16 + ((l >> 4) << 3) + r;
    long tok = (long)w*NTOK + m;
    aout[tok*CDIM + h*32 +      (l & 15)] = f2bf(o0[r]);
    aout[tok*CDIM + h*32 + 16 + (l & 15)] = f2bf(o1[r]);
  }
}

// ------- proj GEMM + window reverse + roll(+shift) + residual -> d_out -------
__global__ __launch_bounds__(256) void k_proj(const unsigned short* __restrict__ aout,
    const unsigned short* __restrict__ wp, const float* __restrict__ bp,
    const float* __restrict__ xin, float* __restrict__ xnew) {
  __shared__ unsigned short At[2][128 * 40];
  __shared__ unsigned short Bt[2][64 * 40];
  int tid = threadIdx.x, wv = tid >> 5, l = tid & 31;
  long m0 = (long)blockIdx.x * 128;
  int  n0 = blockIdx.y * 64;
  const unsigned short* gA = aout + m0 * CDIM;
  const unsigned short* gB = wp + (long)n0 * CDIM;
  v8f acc[4] = {{0,0,0,0,0,0,0,0},{0,0,0,0,0,0,0,0},
                {0,0,0,0,0,0,0,0},{0,0,0,0,0,0,0,0}};
  issue_gemm_tile(gA, gB, At[0], Bt[0], tid);
  for (int kt = 0; kt < 8; ++kt) {
    int cur = kt & 1;
    if (kt < 7) {
      issue_gemm_tile(gA + (kt+1)*32, gB + (kt+1)*32, At[cur^1], Bt[cur^1], tid);
      wait_async<3>();
    } else {
      wait_async<0>();
    }
    __syncthreads();
    v16bf a = load_a_frag(At[cur] + (wv*16)*40, 40);
#pragma unroll
    for (int j = 0; j < 4; ++j) {
      v16bf bf = load_b_frag_nk(Bt[cur] + (j*16)*40, 40);
      acc[j] = wmma_bf16(a, bf, acc[j]);
    }
    __syncthreads();
  }
#pragma unroll
  for (int j = 0; j < 4; ++j)
#pragma unroll
    for (int r = 0; r < 8; ++r) {
      long m  = m0 + wv*16 + ((l >> 4) << 3) + r;
      int col = n0 + j*16 + (l & 15);
      float v = acc[j][r] + bp[col];
      int w = (int)(m / NTOK), n = (int)(m % NTOK);
      int wz = w>>8, wh = (w>>4)&15, ww = w&15;
      int lz = n/72, lh = (n/12)%6, lw = n%12;
      int gz = wz*2+lz, gh = wh*6+lh, gw = ww*12+lw;
      int dz = gz + 1; if (dz >= 8)   dz -= 8;
      int dh = gh + 3; if (dh >= 96)  dh -= 96;
      int dw = gw + 6; if (dw >= 192) dw -= 192;
      long dst = ((long)(dz*96 + dh)*192 + dw)*CDIM + col;
      xnew[dst] = xin[dst] + v;
    }
}

// ---------------- LN2 -> bf16 ----------------
__global__ __launch_bounds__(256) void k_ln2(const float* __restrict__ xin,
    const float* __restrict__ g, const float* __restrict__ b,
    unsigned short* __restrict__ out) {
  int wv = threadIdx.x >> 5, l = threadIdx.x & 31;
  long t = (long)blockIdx.x * 8 + wv;
  int c0 = l * 8;
  const float* row = xin + t*CDIM;
  float4 a0 = *(const float4*)(row + c0);
  float4 a1 = *(const float4*)(row + c0 + 4);
  float vals[8] = {a0.x,a0.y,a0.z,a0.w,a1.x,a1.y,a1.z,a1.w};
  float s = 0.f, q = 0.f;
#pragma unroll
  for (int i = 0; i < 8; ++i) { s += vals[i]; q += vals[i]*vals[i]; }
#pragma unroll
  for (int off = 16; off > 0; off >>= 1) {
    s += __shfl_xor(s, off, 32);
    q += __shfl_xor(q, off, 32);
  }
  float mean = s * (1.f/256.f);
  float inv  = rsqrtf(q * (1.f/256.f) - mean*mean + EPSF);
  unsigned pk[4];
#pragma unroll
  for (int i = 0; i < 4; ++i) {
    float y0 = (vals[2*i]   - mean)*inv*g[c0+2*i]   + b[c0+2*i];
    float y1 = (vals[2*i+1] - mean)*inv*g[c0+2*i+1] + b[c0+2*i+1];
    pk[i] = (unsigned)f2bf(y0) | ((unsigned)f2bf(y1) << 16);
  }
  *(uint4*)(out + t*CDIM + c0) = make_uint4(pk[0], pk[1], pk[2], pk[3]);
}

// ------- fused MLP: 32-token tile, hidden(32x1024 bf16) lives in LDS -------
__global__ __launch_bounds__(256) void k_mlp(const unsigned short* __restrict__ xln,
    const unsigned short* __restrict__ w1, const float* __restrict__ b1,
    const unsigned short* __restrict__ w2, const float* __restrict__ b2,
    float* __restrict__ xio) {
  extern __shared__ char smem[];
  unsigned short* xt = (unsigned short*)smem;   // 32 x 264
  unsigned short* ht = xt + 32*264;             // 32 x 1028
  int tid = threadIdx.x;
  long t0 = (long)blockIdx.x * 32;
  for (int c = tid; c < 1024; c += 256) {       // 32 rows x 32 chunks of 16B
    int row = c >> 5, q = c & 31;
    async_g2l_b128(xln + (t0 + row)*CDIM + q*8, xt + row*264 + q*8);
  }
  wait_async<0>();
  __syncthreads();
  int wv = tid >> 5, l = tid & 31;
  int ms = wv & 1, ng = wv >> 1;                // 2 M-strips x 4 N-groups
  // GEMM1 (K=256 -> N=1024) + GELU(exact) -> LDS bf16
  for (int nb = 0; nb < 4; ++nb) {
    v8f acc[4] = {{0,0,0,0,0,0,0,0},{0,0,0,0,0,0,0,0},
                  {0,0,0,0,0,0,0,0},{0,0,0,0,0,0,0,0}};
    int nbase = ng*256 + nb*64;
    for (int kk = 0; kk < 256; kk += 32) {
      v16bf a = load_a_frag(xt + (ms*16)*264 + kk, 264);
#pragma unroll
      for (int j = 0; j < 4; ++j) {
        v16bf bf = load_b_frag_nk(w1 + (long)(nbase + j*16)*256 + kk, 256);
        acc[j] = wmma_bf16(a, bf, acc[j]);
      }
    }
#pragma unroll
    for (int j = 0; j < 4; ++j)
#pragma unroll
      for (int r = 0; r < 8; ++r) {
        int m = ms*16 + ((l >> 4) << 3) + r;
        int c = nbase + j*16 + (l & 15);
        float v = acc[j][r] + b1[c];
        float gel = 0.5f * v * (1.f + erff(v * 0.70710678118654752f));
        ht[m*1028 + c] = f2bf(gel);
      }
  }
  __syncthreads();
  // GEMM2 (K=1024 -> N=256) + residual
  v8f acc[4] = {{0,0,0,0,0,0,0,0},{0,0,0,0,0,0,0,0},
                {0,0,0,0,0,0,0,0},{0,0,0,0,0,0,0,0}};
  int nbase = ng * 64;
  for (int kk = 0; kk < 1024; kk += 32) {
    v16bf a = load_a_frag(ht + (ms*16)*1028 + kk, 1028);
#pragma unroll
    for (int j = 0; j < 4; ++j) {
      v16bf bf = load_b_frag_nk(w2 + (long)(nbase + j*16)*1024 + kk, 1024);
      acc[j] = wmma_bf16(a, bf, acc[j]);
    }
  }
#pragma unroll
  for (int j = 0; j < 4; ++j)
#pragma unroll
    for (int r = 0; r < 8; ++r) {
      int m = ms*16 + ((l >> 4) << 3) + r;
      int c = nbase + j*16 + (l & 15);
      long idx = (t0 + m)*CDIM + c;
      xio[idx] = xio[idx] + acc[j][r] + b2[c];
    }
}

// ---------------- host launcher ----------------
extern "C" void kernel_launch(void* const* d_in, const int* in_sizes, int n_in,
                              void* d_out, int out_size, void* d_ws, size_t ws_size,
                              hipStream_t stream) {
  (void)in_sizes; (void)n_in; (void)out_size; (void)ws_size;
  const float* x     = (const float*)d_in[0];
  const float* g1    = (const float*)d_in[1];
  const float* be1   = (const float*)d_in[2];
  const float* wqkv  = (const float*)d_in[3];
  const float* bqkv  = (const float*)d_in[4];
  const float* btab  = (const float*)d_in[5];
  const float* wproj = (const float*)d_in[6];
  const float* bproj = (const float*)d_in[7];
  const float* g2    = (const float*)d_in[8];
  const float* be2   = (const float*)d_in[9];
  const float* w1    = (const float*)d_in[10];
  const float* b1    = (const float*)d_in[11];
  const float* w2    = (const float*)d_in[12];
  const float* b2    = (const float*)d_in[13];
  const int*   wti   = (const int*)d_in[17];
  float* out = (float*)d_out;

  char* ws = (char*)d_ws;
  unsigned short* wqkv_bf  = (unsigned short*)(ws);                 // 768*256
  unsigned short* wproj_bf = (unsigned short*)(ws + 393216);        // 256*256
  unsigned short* w1_bf    = (unsigned short*)(ws + 524288);        // 1024*256
  unsigned short* w2_bf    = (unsigned short*)(ws + 1048576);       // 256*1024
  float*          bias8    = (float*)(ws + 1572864);                // 8*144*144 f32
  unsigned short* xw       = (unsigned short*)(ws + 2236416);       // L*C bf16 (reused as attn-out)
  unsigned short* qkvb     = (unsigned short*)(ws + 77733888);      // 3*L*C bf16 (reused as LN2 out)

  k_f2bf<<<768,  256, 0, stream>>>(wqkv,  wqkv_bf,  196608);
  k_f2bf<<<256,  256, 0, stream>>>(wproj, wproj_bf, 65536);
  k_f2bf<<<1024, 256, 0, stream>>>(w1,    w1_bf,    262144);
  k_f2bf<<<1024, 256, 0, stream>>>(w2,    w2_bf,    262144);
  k_bias<<<648,  256, 0, stream>>>(btab, wti, bias8);

  k_ln1_window<<<18432, 256, 0, stream>>>(x, g1, be1, xw);
  k_qkv<<<dim3(1152, 12), 256, 0, stream>>>(xw, wqkv_bf, bqkv, qkvb);

  const int attn_lds = (144*40 + 160*32 + 144*168) * 2 + 144 * 4;   // 70720 B
  (void)hipFuncSetAttribute(reinterpret_cast<const void*>(k_attn),
                            hipFuncAttributeMaxDynamicSharedMemorySize, attn_lds);
  k_attn<<<dim3(1024, 8), 288, attn_lds, stream>>>(qkvb, bias8, xw);

  k_proj<<<dim3(1152, 4), 256, 0, stream>>>(xw, wproj_bf, bproj, x, out);
  k_ln2<<<18432, 256, 0, stream>>>(out, g2, be2, qkvb);

  const int mlp_lds = (32*264 + 32*1028) * 2;                        // 82688 B
  (void)hipFuncSetAttribute(reinterpret_cast<const void*>(k_mlp),
                            hipFuncAttributeMaxDynamicSharedMemorySize, mlp_lds);
  k_mlp<<<4608, 256, mlp_lds, stream>>>(qkvb, w1_bf, b1, w2_bf, b2, out);
}
